// DRAW_44246753083828
// MI455X (gfx1250) — matrix operand order repaired and checked by reading, not compiled, not run on hardware
//
#include <hip/hip_runtime.h>
#include <math.h>

// DRAW (B=128, L=4096, H=1024, Z=256, T=16) on MI455X gfx1250.
// bf16 weights (~100MB -> L2-resident across the 16 steps),
// v_wmma_f32_16x16x32_bf16 GEMMs with fp32 accumulate, fused concat GEMMs,
// async global->LDS (ASYNCcnt) double-buffered B staging with peeled tail.

#define Bc 128
#define Lc 4096
#define Hc 1024
#define Zc 256
#define Tc 16
#define KE 10240   // x(4096) + x_hat(4096) + h_dec(1024) + h_enc(1024)
#define KD 1280    // z(256) + h_dec(1024)
#define GN 4096    // 4*H gate width
#define KP 40      // padded LDS row length (elements) -> 80B row, bank-friendly

typedef __attribute__((ext_vector_type(16))) __bf16       v16bf;
typedef __attribute__((ext_vector_type(8)))  float        v8f;
typedef __attribute__((ext_vector_type(4)))  unsigned int v4u;

union FragBF { v16bf v; v4u q[2]; };

__device__ __forceinline__ unsigned short f2bf(float f) {
  unsigned int u = __float_as_uint(f);
  u += 0x7FFFu + ((u >> 16) & 1u);       // round-to-nearest-even
  return (unsigned short)(u >> 16);
}
__device__ __forceinline__ float sigf(float x) { return 1.0f / (1.0f + __expf(-x)); }

// ---------------------------------------------------------------------------
// WMMA GEMM: Cout[128][N] = A_bf16[128][K] * Bw_bf16[N][K]^T + bias
//            (+ Cin accumulate if Cin != null; Cin may alias Cout).
// Block = 128 threads = 4 waves; wave w owns M rows [64*blockIdx.y + 16w, +16).
// Grid = (N/64, 128/64). K % 32 == 0, K >= 64.
// B tile (32K x 64N) double-buffered in LDS via GLOBAL_LOAD_ASYNC_TO_LDS_B128;
// steady-state loop is branch-free (tail chunk peeled).
// ---------------------------------------------------------------------------
__global__ __launch_bounds__(128)
void k_gemm_bf16(const unsigned short* __restrict__ A,
                 const unsigned short* __restrict__ Bw,
                 const float* __restrict__ bias,
                 const float* __restrict__ Cin,
                 float* __restrict__ Cout,
                 int K, int N)
{
  __shared__ unsigned short ldsB[2 * 64 * KP];  // two 64-row x 32-elem tiles, padded

  const int nBase = blockIdx.x * 64;
  const int lane  = threadIdx.x & 31;
  const int wave  = threadIdx.x >> 5;           // 0..3
  const int m0    = blockIdx.y * 64 + wave * 16;

  // A fragment addressing (ISA 16-bit A 16x32 layout):
  // lanes 0-15: row M=lane,     K = {0..7, 16..23}
  // lanes 16-31: row M=lane-16, K = {8..15, 24..31}
  const int aRow  = m0 + (lane & 15);
  const int aKoff = (lane >> 4) * 8;
  const unsigned short* aBase = A + (size_t)aRow * K + aKoff;

  // B fragment addressing (ISA 16-bit B 32x16 layout):
  // lane -> column N (mod 16); lanes 0-15 hold K {0..7, 16..23}, lanes 16-31 hold K {8..15, 24..31}
  const int bCol   = lane & 15;
  const int bKhalf = (lane >> 4) * 8;

  // async staging: thread t moves 32B of row tn (two b128 ops)
  const int tn  = threadIdx.x >> 1;             // 0..63
  const int tkB = (threadIdx.x & 1) * 16;       // element slice 0 or 16
  const unsigned short* bBase = Bw + (size_t)(nBase + tn) * K + tkB;
  // ISA 10.2: flat LDS addresses carry the LDS byte offset in addr[31:0]
  const unsigned ldsStage0 = (unsigned)(uintptr_t)&ldsB[tn * KP + tkB];

  v8f acc[4] = {};

  auto stage = [&](int k0, int buf) {
    unsigned dst = ldsStage0 + (unsigned)(buf * 64 * KP * 2);
    unsigned long long src = (unsigned long long)(uintptr_t)(bBase + k0);
    asm volatile(
        "global_load_async_to_lds_b128 %0, %1, off\n\t"
        "global_load_async_to_lds_b128 %2, %3, off"
        :: "v"(dst), "v"(src), "v"(dst + 16u), "v"(src + 16ull)
        : "memory");
  };

  auto compute = [&](int k0, int buf) {
    FragBF a;
    a.q[0] = *reinterpret_cast<const v4u*>(aBase + k0);
    a.q[1] = *reinterpret_cast<const v4u*>(aBase + k0 + 16);
    const unsigned short* bufBase = &ldsB[buf * 64 * KP];
#pragma unroll
    for (int j = 0; j < 4; ++j) {
      FragBF b;
      const unsigned short* bp = bufBase + (j * 16 + bCol) * KP + bKhalf;
      b.q[0] = *reinterpret_cast<const v4u*>(bp);
      b.q[1] = *reinterpret_cast<const v4u*>(bp + 16);
      acc[j] = __builtin_amdgcn_wmma_f32_16x16x32_bf16(
          false, a.v, false, b.v, (short)0, acc[j], false, false);
    }
  };

  // software pipeline: branch-free steady state, peeled tail chunk
  stage(0, 0);
  int buf = 0;
  int k0 = 0;
  for (; k0 + 32 < K; k0 += 32) {
    stage(k0 + 32, buf ^ 1);                          // prefetch next chunk
    asm volatile("s_wait_asynccnt 0x2" ::: "memory"); // current chunk landed
    __syncthreads();
    compute(k0, buf);
    __syncthreads();
    buf ^= 1;
  }
  asm volatile("s_wait_asynccnt 0x0" ::: "memory");   // drain
  __syncthreads();
  compute(k0, buf);

  // D layout: VGPR r -> lanes 0-15: M=r, lanes 16-31: M=r+8; N = lane&15
  const int outN0 = nBase + (lane & 15);
  const int outM0 = m0 + (lane >> 4) * 8;
  if (Cin) {
#pragma unroll
    for (int j = 0; j < 4; ++j) {
      const int n = outN0 + j * 16;
      const float bv = bias[n];
#pragma unroll
      for (int rr = 0; rr < 8; ++rr) {
        const size_t idx = (size_t)(outM0 + rr) * N + n;
        Cout[idx] = acc[j][rr] + bv + Cin[idx];
      }
    }
  } else {
#pragma unroll
    for (int j = 0; j < 4; ++j) {
      const int n = outN0 + j * 16;
      const float bv = bias[n];
#pragma unroll
      for (int rr = 0; rr < 8; ++rr) {
        const size_t idx = (size_t)(outM0 + rr) * N + n;
        Cout[idx] = acc[j][rr] + bv;
      }
    }
  }
}

// ------------------------- elementwise helpers -----------------------------

__global__ void k_cvt(const float* __restrict__ src, unsigned short* __restrict__ dst,
                      int rows, int cols, int dstStride, int dstOff)
{
  int i = blockIdx.x * blockDim.x + threadIdx.x;
  if (i >= rows * cols) return;
  int r = i / cols, c = i - r * cols;
  dst[(size_t)r * dstStride + dstOff + c] = f2bf(src[i]);
}

__global__ void k_add(const float* __restrict__ a, const float* __restrict__ b,
                      float* __restrict__ dst, int n)
{
  int i = blockIdx.x * blockDim.x + threadIdx.x;
  if (i >= n) return;
  dst[i] = b ? (a[i] + b[i]) : a[i];
}

// init: c_t = broadcast(c_0); r[:, 0:L] = bf16(x)
__global__ void k_init(const float* __restrict__ x, const float* __restrict__ c0,
                       float* __restrict__ ct, unsigned short* __restrict__ r)
{
  int i = blockIdx.x * blockDim.x + threadIdx.x;
  if (i >= Bc * Lc) return;
  int b = i >> 12, l = i & (Lc - 1);
  ct[i] = c0[l];
  r[(size_t)b * KE + l] = f2bf(x[i]);
}

__global__ void k_init_h(const float* __restrict__ h0e, const float* __restrict__ h0d,
                         unsigned short* __restrict__ r, unsigned short* __restrict__ zd,
                         float* __restrict__ cenc, float* __restrict__ cdec)
{
  int i = blockIdx.x * blockDim.x + threadIdx.x;
  if (i >= Bc * Hc) return;
  int b = i >> 10, j = i & (Hc - 1);
  unsigned short hd = f2bf(h0d[j]);
  r[(size_t)b * KE + 2 * Lc + j]      = hd;            // h_dec slot in r
  r[(size_t)b * KE + 2 * Lc + Hc + j] = f2bf(h0e[j]);  // h_enc slot in r
  zd[(size_t)b * KD + Zc + j]         = hd;            // h_dec slot in decoder A
  cenc[i] = 0.0f;
  cdec[i] = 0.0f;
}

// per-step: r[:, L:2L] = bf16(x - sigmoid(c_t))
__global__ void k_prestep(const float* __restrict__ x, const float* __restrict__ ct,
                          unsigned short* __restrict__ r)
{
  int i = blockIdx.x * blockDim.x + threadIdx.x;
  if (i >= Bc * Lc) return;
  int b = i >> 12, l = i & (Lc - 1);
  r[(size_t)b * KE + Lc + l] = f2bf(x[i] - sigf(ct[i]));
}

// LSTM cell (gate order i,f,g,o); writes new h (bf16) to up to 3 consumer slots
__global__ void k_cell(const float* __restrict__ gates, float* __restrict__ c,
                       unsigned short* __restrict__ h1, int st1,
                       unsigned short* __restrict__ h2, int st2,
                       unsigned short* __restrict__ h3, int st3)
{
  int i = blockIdx.x * blockDim.x + threadIdx.x;
  if (i >= Bc * Hc) return;
  int b = i >> 10, j = i & (Hc - 1);
  const float* g = gates + (size_t)b * GN;
  float ig = sigf(g[j]);
  float fg = sigf(g[Hc + j]);
  float gg = tanhf(g[2 * Hc + j]);
  float og = sigf(g[3 * Hc + j]);
  float cn = fg * c[i] + ig * gg;
  c[i] = cn;
  unsigned short hb = f2bf(og * tanhf(cn));
  h1[(size_t)b * st1 + j] = hb;
  h2[(size_t)b * st2 + j] = hb;
  if (h3) h3[(size_t)b * st3 + j] = hb;
}

// z = eps_t * exp(logsigma) + mu  (musig = [mu | logsigma], 512 wide)
__global__ void k_reparam(const float* __restrict__ musig, const float* __restrict__ epsT,
                          unsigned short* __restrict__ zd)
{
  int i = blockIdx.x * blockDim.x + threadIdx.x;
  if (i >= Bc * Zc) return;
  int b = i >> 8, z = i & (Zc - 1);
  float mu = musig[(size_t)b * 512 + z];
  float ls = musig[(size_t)b * 512 + Zc + z];
  zd[(size_t)b * KD + z] = f2bf(epsT[i] * __expf(ls) + mu);
}

__global__ void k_final(const float* __restrict__ ct, float* __restrict__ out)
{
  int i = blockIdx.x * blockDim.x + threadIdx.x;
  if (i >= Bc * Lc) return;
  out[i] = sigf(ct[i]);
}

// ---------------------------------------------------------------------------

extern "C" void kernel_launch(void* const* d_in, const int* in_sizes, int n_in,
                              void* d_out, int out_size, void* d_ws, size_t ws_size,
                              hipStream_t stream) {
  const float* x     = (const float*)d_in[0];
  const float* eps   = (const float*)d_in[1];
  const float* c0    = (const float*)d_in[2];
  const float* h0e   = (const float*)d_in[3];
  const float* h0d   = (const float*)d_in[4];
  const float* WihE  = (const float*)d_in[5];
  const float* bihE  = (const float*)d_in[6];
  const float* WhhE  = (const float*)d_in[7];
  const float* bhhE  = (const float*)d_in[8];
  const float* Wmu   = (const float*)d_in[9];
  const float* bmu   = (const float*)d_in[10];
  const float* Wsig  = (const float*)d_in[11];
  const float* bsig  = (const float*)d_in[12];
  const float* WihD  = (const float*)d_in[13];
  const float* bihD  = (const float*)d_in[14];
  const float* WhhD  = (const float*)d_in[15];
  const float* bhhD  = (const float*)d_in[16];
  const float* Wwr   = (const float*)d_in[17];
  const float* bwr   = (const float*)d_in[18];
  float* out = (float*)d_out;

  char* ws = (char*)d_ws;
  size_t cur = 0;
  auto alloc = [&](size_t bytes) -> void* {
    void* p = ws + cur;
    cur = (cur + bytes + 255) & ~(size_t)255;
    return p;
  };

  // bf16 weight pool (~100 MB -> fits MI455X 192MB L2 across the 16-step loop)
  unsigned short* wEnc = (unsigned short*)alloc((size_t)GN * KE * 2);  // [4096][10240]
  unsigned short* wMs  = (unsigned short*)alloc((size_t)512 * Hc * 2); // [512][1024]
  unsigned short* wDec = (unsigned short*)alloc((size_t)GN * KD * 2);  // [4096][1280]
  unsigned short* wWr  = (unsigned short*)alloc((size_t)Lc * Hc * 2);  // [4096][1024]
  float* biasE  = (float*)alloc(GN * 4);
  float* biasMS = (float*)alloc(512 * 4);
  float* biasD  = (float*)alloc(GN * 4);
  // activations
  float*          ct    = (float*)alloc((size_t)Bc * Lc * 4);
  unsigned short* r     = (unsigned short*)alloc((size_t)Bc * KE * 2);
  float*          gates = (float*)alloc((size_t)Bc * GN * 4);   // shared enc/dec
  float*          cenc  = (float*)alloc((size_t)Bc * Hc * 4);
  float*          cdec  = (float*)alloc((size_t)Bc * Hc * 4);
  unsigned short* msA   = (unsigned short*)alloc((size_t)Bc * Hc * 2);
  float*          musig = (float*)alloc((size_t)Bc * 512 * 4);
  unsigned short* zd    = (unsigned short*)alloc((size_t)Bc * KD * 2);
  unsigned short* hdA   = (unsigned short*)alloc((size_t)Bc * Hc * 2);
  (void)ws_size; (void)in_sizes; (void)n_in; (void)out_size;

  const int TB = 256;
  auto blocks = [](long long n) { return (unsigned)((n + 255) / 256); };

  // --- repack weights to bf16 (fused layouts) ---
  k_cvt<<<blocks((long long)GN * (2 * Lc + Hc)), TB, 0, stream>>>(WihE, wEnc, GN, 2 * Lc + Hc, KE, 0);
  k_cvt<<<blocks((long long)GN * Hc), TB, 0, stream>>>(WhhE, wEnc, GN, Hc, KE, 2 * Lc + Hc);
  k_cvt<<<blocks((long long)Zc * Hc), TB, 0, stream>>>(Wmu, wMs, Zc, Hc, Hc, 0);
  k_cvt<<<blocks((long long)Zc * Hc), TB, 0, stream>>>(Wsig, wMs + (size_t)Zc * Hc, Zc, Hc, Hc, 0);
  k_cvt<<<blocks((long long)GN * Zc), TB, 0, stream>>>(WihD, wDec, GN, Zc, KD, 0);
  k_cvt<<<blocks((long long)GN * Hc), TB, 0, stream>>>(WhhD, wDec, GN, Hc, KD, Zc);
  k_cvt<<<blocks((long long)Lc * Hc), TB, 0, stream>>>(Wwr, wWr, Lc, Hc, Hc, 0);
  // fused biases
  k_add<<<blocks(GN), TB, 0, stream>>>(bihE, bhhE, biasE, GN);
  k_add<<<blocks(Zc), TB, 0, stream>>>(bmu, nullptr, biasMS, Zc);
  k_add<<<blocks(Zc), TB, 0, stream>>>(bsig, nullptr, biasMS + Zc, Zc);
  k_add<<<blocks(GN), TB, 0, stream>>>(bihD, bhhD, biasD, GN);

  // --- initial state ---
  k_init<<<blocks((long long)Bc * Lc), TB, 0, stream>>>(x, c0, ct, r);
  k_init_h<<<blocks((long long)Bc * Hc), TB, 0, stream>>>(h0e, h0d, r, zd, cenc, cdec);

  const dim3 gGate(GN / 64, Bc / 64);   // 64 x 2 = 128 blocks
  const dim3 gMs(512 / 64, Bc / 64);    //  8 x 2
  const dim3 gWr(Lc / 64, Bc / 64);     // 64 x 2

  // --- 16 sequential DRAW steps ---
  for (int t = 0; t < Tc; ++t) {
    // x_hat into r[:, L:2L]
    k_prestep<<<blocks((long long)Bc * Lc), TB, 0, stream>>>(x, ct, r);
    // encoder gates: [x, x_hat, h_dec, h_enc] (K=10240) x W_enc^T -> [128][4096]
    k_gemm_bf16<<<gGate, 128, 0, stream>>>(r, wEnc, biasE, nullptr, gates, KE, GN);
    // encoder cell -> h_enc bf16 into msA and r[:, 9216:10240] (next step)
    k_cell<<<blocks((long long)Bc * Hc), TB, 0, stream>>>(
        gates, cenc, msA, Hc, r + (size_t)(2 * Lc + Hc), KE, nullptr, 0);
    // [mu | logsigma]: h_enc (K=1024) x W_ms^T -> [128][512]
    k_gemm_bf16<<<gMs, 128, 0, stream>>>(msA, wMs, biasMS, nullptr, musig, Hc, 512);
    // z into zd[:, 0:256]
    k_reparam<<<blocks((long long)Bc * Zc), TB, 0, stream>>>(
        musig, eps + (size_t)t * Bc * Zc, zd);
    // decoder gates: [z, h_dec] (K=1280) x W_dec^T -> [128][4096]
    k_gemm_bf16<<<gGate, 128, 0, stream>>>(zd, wDec, biasD, nullptr, gates, KD, GN);
    // decoder cell -> h_dec bf16 into write-A, r[:, 8192:9216], zd[:, 256:1280]
    k_cell<<<blocks((long long)Bc * Hc), TB, 0, stream>>>(
        gates, cdec, hdA, Hc, r + (size_t)(2 * Lc), KE, zd + Zc, KD);
    // canvas: c_t += h_dec x W_write^T + b_write (in-place accumulate)
    k_gemm_bf16<<<gWr, 128, 0, stream>>>(hdA, wWr, bwr, ct, ct, Hc, Lc);
  }

  k_final<<<blocks((long long)Bc * Lc), TB, 0, stream>>>(ct, out);
}
